// SpaceSelfAttentionModality_2061584302930
// MI455X (gfx1250) — matrix operand order repaired and checked by reading, not compiled, not run on hardware
//
#include <hip/hip_runtime.h>
#include <hip/hip_bf16.h>

typedef _Float16 f16;
typedef __attribute__((ext_vector_type(16))) _Float16 v16h;
typedef __attribute__((ext_vector_type(8)))  _Float16 v8h;
typedef __attribute__((ext_vector_type(8)))  float    v8f;

#define NEG_INF (-__builtin_inff())

// Problem constants
#define BB 4
#define TT 8
#define SS 512
#define DD 1024
#define HH 16
#define HD 64
#define NBT 32              // B*T
#define MROWS (NBT * SS)    // 16384 total rows

__device__ __forceinline__ v16h cat8(v8h lo, v8h hi) {
    return __builtin_shufflevector(lo, hi, 0,1,2,3,4,5,6,7,8,9,10,11,12,13,14,15);
}

__device__ __forceinline__ v8f wmma_f16(v16h a, v16h b, v8f c) {
    return __builtin_amdgcn_wmma_f32_16x16x32_f16(false, a, false, b, (short)0, c, false, false);
}

// ---------------- register-blocked 64x64 GEMM core ----------------
// Wave computes a 64x64 output block as 4x4 WMMA tiles (128 acc VGPRs).
// A row-major [.,lda], Bt holds B transposed: Bt[n][k] so the per-lane
// B-fragment (col = lane&15, K = (lane>>4)*16+0..15) is one contiguous v16h.
// Per K=32 step: 8+8 b128 loads feed 16 WMMAs (1 load per WMMA).
__device__ __forceinline__ void gemm_wave_4x4(const f16* __restrict__ A, int lda,
                                              const f16* __restrict__ Bt, int ldb,
                                              int K, int lane, v8f acc[4][4]) {
    const f16* ar[4];
    const f16* br[4];
#pragma unroll
    for (int i = 0; i < 4; ++i)
        ar[i] = A + (size_t)(i * 16 + (lane & 15)) * lda + ((lane >> 4) << 4);
#pragma unroll
    for (int j = 0; j < 4; ++j)
        br[j] = Bt + (size_t)(j * 16 + (lane & 15)) * ldb + ((lane >> 4) << 4);
    for (int kk = 0; kk < K; kk += 32) {
        v16h a[4];
#pragma unroll
        for (int i = 0; i < 4; ++i)
            a[i] = cat8(*(const v8h*)(ar[i] + kk), *(const v8h*)(ar[i] + kk + 8));
#pragma unroll
        for (int j = 0; j < 4; ++j) {
            v16h b = *(const v16h*)(br[j] + kk);
#pragma unroll
            for (int i = 0; i < 4; ++i)
                acc[i][j] = wmma_f16(a[i], b, acc[i][j]);
        }
    }
}

// ---------------- conversion kernels ----------------

__global__ void cvt_x(const float* __restrict__ x, f16* __restrict__ xh, int n) {
    int i = blockIdx.x * blockDim.x + threadIdx.x;
    if (i < n) xh[i] = (f16)x[i];
}

// wq/wk/wv: [D, H*HD] row-major -> wqkvT[m][o][d] (transposed)
// wo: [H*HD, D] row-major      -> woT[n][k]       (transposed)
__global__ void cvt_w(const float* __restrict__ wq, const float* __restrict__ wk,
                      const float* __restrict__ wv, const float* __restrict__ wo,
                      f16* __restrict__ wqkvT, f16* __restrict__ woT) {
    int i = blockIdx.x * blockDim.x + threadIdx.x;   // 0 .. 4*1024*1024-1
    int m = i >> 20;
    int r = i & ((1 << 20) - 1);
    int a = r >> 10;      // output column (o) / output dim (n)
    int d = r & 1023;     // input dim (d) / inner (k)
    if (m < 3) {
        const float* w = (m == 0) ? wq : (m == 1) ? wk : wv;
        wqkvT[((size_t)m << 20) + ((size_t)a << 10) + d] = (f16)w[(size_t)d * 1024 + a];
    } else {
        woT[((size_t)a << 10) + d] = (f16)wo[(size_t)d * 1024 + a];
    }
}

// ---------------- fused QKV projection GEMM ----------------
// X[16384,1024] * [Wq|Wk|Wv][1024,3072]; one 64x64 block per wave.
// q scaled by 1/8 and biased; k biased; v stored transposed [bt][h][HD][S].
__global__ __launch_bounds__(256) void qkv_gemm(
    const f16* __restrict__ xh, const f16* __restrict__ wqkvT,
    const float* __restrict__ bq, const float* __restrict__ bk, const float* __restrict__ bv,
    f16* __restrict__ qbuf, f16* __restrict__ kbuf, f16* __restrict__ vT) {
    int wave = (blockIdx.x * blockDim.x + threadIdx.x) >> 5;  // 0..12287
    int lane = threadIdx.x & 31;
    int tm = wave / 48;           // 0..255 (64-row blocks)
    int tn = wave % 48;           // 0..47  (64-col blocks over 3072)
    const f16* A  = xh    + (size_t)(tm * 64) * DD;
    const f16* Bt = wqkvT + (size_t)(tn * 64) * DD;
    v8f acc[4][4];
#pragma unroll
    for (int i = 0; i < 4; ++i)
#pragma unroll
        for (int j = 0; j < 4; ++j) acc[i][j] = (v8f){};
    gemm_wave_4x4(A, DD, Bt, DD, DD, lane, acc);

    int m = (tn * 64) >> 10;              // 0=q 1=k 2=v (uniform per wave)
    int h = ((tn * 64) & 1023) >> 6;      // head (uniform per wave)
    const float* bias = (m == 0) ? bq : (m == 1) ? bk : bv;
    float scale = (m == 0) ? 0.125f : 1.0f;
    f16* qk = (m == 0) ? qbuf : kbuf;
#pragma unroll
    for (int j = 0; j < 4; ++j) {
        int c = j * 16 + (lane & 15);     // channel within head
        float bsv = bias[h * 64 + c];
#pragma unroll
        for (int i = 0; i < 4; ++i) {
            int sg0 = tm * 64 + i * 16 + ((lane >> 4) << 3);  // first row for this lane
            int bt = sg0 >> 9, s0 = sg0 & 511;                // no bt crossing in a tile
            if (m == 2) {
                // v transposed: rows r are consecutive s -> one packed b128 store
                v8h pk;
#pragma unroll
                for (int r = 0; r < 8; ++r) pk[r] = (f16)(acc[i][j][r] + bsv);
                *(v8h*)(vT + (((size_t)(bt * HH + h) << 6) + c) * SS + s0) = pk;
            } else {
#pragma unroll
                for (int r = 0; r < 8; ++r) {
                    f16 hv = (f16)((acc[i][j][r] + bsv) * scale);
                    qk[(((size_t)(bt * HH + h) << 9) + (s0 + r)) * HD + c] = hv;
                }
            }
        }
    }
}

// ---------------- flash attention per (bt, h, 16-query tile) ----------------
__global__ __launch_bounds__(256) void attn(
    const f16* __restrict__ qbuf, const f16* __restrict__ kbuf, const f16* __restrict__ vT,
    const int* __restrict__ mods, f16* __restrict__ abuf) {
    __shared__ __align__(32) f16 lds[8][16 * 32];   // per-wave P staging tile
    int wslot = threadIdx.x >> 5;
    int wave  = (blockIdx.x * blockDim.x + threadIdx.x) >> 5;  // 0..16383
    int lane  = threadIdx.x & 31;
    int qt = wave & 31;
    int h  = (wave >> 5) & 15;
    int bt = wave >> 9;
    int q0 = qt << 4;

    const f16* qbase = qbuf + (size_t)(bt * HH + h) * (SS * HD);
    const f16* kbase = kbuf + (size_t)(bt * HH + h) * (SS * HD);
    const f16* vbase = vT   + (size_t)(bt * HH + h) * (HD * SS);
    f16* myp = &lds[wslot][0];

    // Q A-fragments (row = lane&15, K = (lane>>4)*16+0..15), HD=64 -> two K=32 frags
    const f16* qrow = qbase + (size_t)(q0 + (lane & 15)) * HD + ((lane >> 4) << 4);
    v16h a0 = cat8(*(const v8h*)(qrow),      *(const v8h*)(qrow + 8));
    v16h a1 = cat8(*(const v8h*)(qrow + 32), *(const v8h*)(qrow + 40));

    int  mq[8];
    bool lat[8];
    int mbase = q0 + ((lane >> 4) << 3);
#pragma unroll
    for (int r = 0; r < 8; ++r) { int qi = mbase + r; mq[r] = mods[qi]; lat[r] = (qi < 16); }

    float mrow[8], lrow[8];
#pragma unroll
    for (int r = 0; r < 8; ++r) { mrow[r] = NEG_INF; lrow[r] = 0.0f; }
    v8f o0 = {}, o1 = {}, o2 = {}, o3 = {};

    for (int kb = 0; kb < SS; kb += 32) {
        // ---- scores for 2 x 16 keys ----
        const f16* kr0 = kbase + (size_t)(kb + (lane & 15)) * HD + ((lane >> 4) << 4);
        const f16* kr1 = kr0 + 16 * HD;
        v8f st0 = {}, st1 = {};
        st0 = wmma_f16(a0, *(const v16h*)(kr0), st0);
        st0 = wmma_f16(a1, *(const v16h*)(kr0 + 32), st0);
        st1 = wmma_f16(a0, *(const v16h*)(kr1), st1);
        st1 = wmma_f16(a1, *(const v16h*)(kr1 + 32), st1);
        int mk0 = mods[kb + (lane & 15)];
        int mk1 = mods[kb + 16 + (lane & 15)];

        // ---- mask + online softmax (rows across 16-lane halves) ----
        float p0a[8], p1a[8];
#pragma unroll
        for (int r = 0; r < 8; ++r) {
            float v0 = (lat[r] || (mq[r] == mk0)) ? st0[r] : NEG_INF;
            float v1 = (lat[r] || (mq[r] == mk1)) ? st1[r] : NEG_INF;
            float mx = fmaxf(v0, v1);
#pragma unroll
            for (int dd = 1; dd < 16; dd <<= 1) mx = fmaxf(mx, __shfl_xor(mx, dd, 16));
            float mn = fmaxf(mrow[r], mx);
            float corr = (mn == NEG_INF) ? 1.0f : __expf(mrow[r] - mn);
            float p0 = (v0 == NEG_INF) ? 0.0f : __expf(v0 - mn);
            float p1 = (v1 == NEG_INF) ? 0.0f : __expf(v1 - mn);
            float rs = p0 + p1;
#pragma unroll
            for (int dd = 1; dd < 16; dd <<= 1) rs += __shfl_xor(rs, dd, 16);
            lrow[r] = lrow[r] * corr + rs;
            mrow[r] = mn;
            o0[r] *= corr; o1[r] *= corr; o2[r] *= corr; o3[r] *= corr;
            p0a[r] = p0;  p1a[r] = p1;
        }

        // ---- re-swizzle P (C/D layout -> A layout) through LDS ----
#pragma unroll
        for (int r = 0; r < 8; ++r) {
            int M = r + ((lane >> 4) << 3);
            myp[M * 32 + (lane & 15)]      = (f16)p0a[r];
            myp[M * 32 + 16 + (lane & 15)] = (f16)p1a[r];
        }
        asm volatile("s_wait_dscnt 0" ::: "memory");
        const f16* prow = myp + (lane & 15) * 32 + ((lane >> 4) << 4);
        v16h ap = cat8(*(const v8h*)(prow), *(const v8h*)(prow + 8));

        // ---- P(16x32) * V(32x64): B cols contiguous thanks to vT ----
        const f16* vr = vbase + (size_t)(lane & 15) * SS + kb + ((lane >> 4) << 4);
        o0 = wmma_f16(ap, *(const v16h*)(vr),            o0);
        o1 = wmma_f16(ap, *(const v16h*)(vr + 16 * SS),  o1);
        o2 = wmma_f16(ap, *(const v16h*)(vr + 32 * SS),  o2);
        o3 = wmma_f16(ap, *(const v16h*)(vr + 48 * SS),  o3);
    }

    // ---- normalize and store concat-head layout [row][h*64+c] ----
#pragma unroll
    for (int r = 0; r < 8; ++r) {
        float inv = 1.0f / lrow[r];
        int sg = bt * SS + q0 + r + ((lane >> 4) << 3);
        size_t base = (size_t)sg * DD + (size_t)h * HD + (lane & 15);
        abuf[base]      = (f16)(o0[r] * inv);
        abuf[base + 16] = (f16)(o1[r] * inv);
        abuf[base + 32] = (f16)(o2[r] * inv);
        abuf[base + 48] = (f16)(o3[r] * inv);
    }
}

// ---------------- output projection GEMM + bias (fp32 out) ----------------
__global__ __launch_bounds__(256) void out_gemm(
    const f16* __restrict__ abuf, const f16* __restrict__ woT,
    const float* __restrict__ bo, float* __restrict__ y) {
    int wave = (blockIdx.x * blockDim.x + threadIdx.x) >> 5;  // 0..4095
    int lane = threadIdx.x & 31;
    int tm = wave >> 4;          // 256 row blocks of 64
    int tn = wave & 15;          // 16 col blocks of 64
    const f16* A  = abuf + (size_t)(tm * 64) * DD;
    const f16* Bt = woT  + (size_t)(tn * 64) * DD;
    v8f acc[4][4];
#pragma unroll
    for (int i = 0; i < 4; ++i)
#pragma unroll
        for (int j = 0; j < 4; ++j) acc[i][j] = (v8f){};
    gemm_wave_4x4(A, DD, Bt, DD, DD, lane, acc);

#pragma unroll
    for (int j = 0; j < 4; ++j) {
        int col = tn * 64 + j * 16 + (lane & 15);
        float b = bo[col];
#pragma unroll
        for (int i = 0; i < 4; ++i) {
#pragma unroll
            for (int r = 0; r < 8; ++r) {
                int row = tm * 64 + i * 16 + r + ((lane >> 4) << 3);
                y[(size_t)row * DD + col] = acc[i][j][r] + b;
            }
        }
    }
}

// ---------------- host launcher ----------------
extern "C" void kernel_launch(void* const* d_in, const int* in_sizes, int n_in,
                              void* d_out, int out_size, void* d_ws, size_t ws_size,
                              hipStream_t stream) {
    const float* x    = (const float*)d_in[0];
    const int*   mods = (const int*)d_in[1];
    const float* wq   = (const float*)d_in[2];
    const float* bq   = (const float*)d_in[3];
    const float* wk   = (const float*)d_in[4];
    const float* bk   = (const float*)d_in[5];
    const float* wv   = (const float*)d_in[6];
    const float* bv   = (const float*)d_in[7];
    const float* wo   = (const float*)d_in[8];
    const float* bo   = (const float*)d_in[9];
    float* y = (float*)d_out;

    char* ws = (char*)d_ws;
    size_t off = 0;
    auto alloc = [&](size_t elems) { f16* p = (f16*)(ws + off); off += ((elems * sizeof(f16) + 255) & ~(size_t)255); return p; };
    f16* xh    = alloc((size_t)MROWS * DD);        // 32 MB
    f16* wqkvT = alloc((size_t)3 * DD * DD);       // 6 MB
    f16* woT   = alloc((size_t)DD * DD);           // 2 MB
    f16* qb    = alloc((size_t)NBT * HH * SS * HD);
    f16* kb_   = alloc((size_t)NBT * HH * SS * HD);
    f16* vT    = alloc((size_t)NBT * HH * HD * SS);
    f16* ab    = alloc((size_t)MROWS * DD);

    cvt_x<<<(MROWS * DD) / 256, 256, 0, stream>>>(x, xh, MROWS * DD);
    cvt_w<<<(4 * 1024 * 1024) / 256, 256, 0, stream>>>(wq, wk, wv, wo, wqkvT, woT);
    // 256 row-blocks x 48 col-blocks, 8 waves (64x64 blocks) per 256-thread block
    qkv_gemm<<<(256 * 48) / 8, 256, 0, stream>>>(xh, wqkvT, bq, bk, bv, qb, kb_, vT);
    // 32 bt x 16 h x 32 qtiles = 16384 waves
    attn<<<16384 / 8, 256, 0, stream>>>(qb, kb_, vT, mods, ab);
    // 256 x 16 blocks of 64x64
    out_gemm<<<(256 * 16) / 8, 256, 0, stream>>>(ab, woT, bo, y);
}
